// InvertedResidualBlockME_10256381903469
// MI455X (gfx1250) — matrix operand order repaired and checked by reading, not compiled, not run on hardware
//
#include <hip/hip_runtime.h>

// ---------------------------------------------------------------------------
// Inverted residual Minkowski block, MI455X (gfx1250, wave32, WMMA).
//   1. zero BN stats
//   2. GEMM1 (bf16 WMMA, f32 accum, B via LDS): h1 = x @ W1        [N,384]
//   3. column stats of h1  -> scale1/shift1
//   4. depthwise gather:  h2[i] = sum_k m[i,k]*relu6(bn1(h1[idx[i,k]]))*w2[k]
//   5. column stats of h2  -> scale2/shift2
//   6. GEMM3 (bf16 WMMA): yraw = relu6(bn2(h2)) @ W3               [N,64]
//   7. column stats of yraw -> scale3/shift3
//   8. out = bn3(yraw) + x
// Memory-bound (~4.3 GB HBM @ 23.3 TB/s). Wave tile = 16x64 (4 WMMA accums)
// so A is loaded once per k-step; weights staged in LDS as bf16 column-major.
// ---------------------------------------------------------------------------

#define CIN   64
#define HID   384
#define COUT  64
#define KVOL  9
#define BN_EPS 1e-5f

typedef __bf16 bf16_t;
typedef bf16_t v16bf __attribute__((ext_vector_type(16)));
typedef bf16_t v8bf  __attribute__((ext_vector_type(8)));
typedef float  v8f   __attribute__((ext_vector_type(8)));

// ---------------- WMMA A-fragment loaders (ISA 16-bit layout, wave32) ------
// A 16x32 bf16 tile from row-major f32. lane<16: row=lane, K={0..7,16..23};
// lane>=16: row=lane-16, K={8..15,24..31}. v16bf elem j -> VGPR j/2 half j%2.
__device__ __forceinline__ v16bf load_a_frag(const float* __restrict__ A, int lda,
                                             int row0, int k0, int lane) {
  const int m  = row0 + (lane & 15);
  const int kb = (lane >> 4) << 3;           // 0 or 8
  const float4* p0 = reinterpret_cast<const float4*>(A + (size_t)m * lda + k0 + kb);
  const float4* p1 = reinterpret_cast<const float4*>(A + (size_t)m * lda + k0 + 16 + kb);
  float4 a0 = p0[0], a1 = p0[1], b0 = p1[0], b1 = p1[1];
  v16bf a;
  a[0]=(bf16_t)a0.x; a[1]=(bf16_t)a0.y; a[2]=(bf16_t)a0.z; a[3]=(bf16_t)a0.w;
  a[4]=(bf16_t)a1.x; a[5]=(bf16_t)a1.y; a[6]=(bf16_t)a1.z; a[7]=(bf16_t)a1.w;
  a[8]=(bf16_t)b0.x; a[9]=(bf16_t)b0.y; a[10]=(bf16_t)b0.z; a[11]=(bf16_t)b0.w;
  a[12]=(bf16_t)b1.x; a[13]=(bf16_t)b1.y; a[14]=(bf16_t)b1.z; a[15]=(bf16_t)b1.w;
  return a;
}

// Same A tile with BN affine (indexed by K) + relu6 before bf16 conversion.
__device__ __forceinline__ v16bf load_a_frag_bn(const float* __restrict__ A, int lda,
                                                const float* __restrict__ sc,
                                                const float* __restrict__ sh,
                                                int row0, int k0, int lane) {
  const int m  = row0 + (lane & 15);
  const int kb = (lane >> 4) << 3;
  v16bf a;
#pragma unroll
  for (int half = 0; half < 2; ++half) {
    const int kbeg = k0 + half * 16 + kb;
    const float4* p  = reinterpret_cast<const float4*>(A + (size_t)m * lda + kbeg);
    const float4* ps = reinterpret_cast<const float4*>(sc + kbeg);
    const float4* pb = reinterpret_cast<const float4*>(sh + kbeg);
#pragma unroll
    for (int q = 0; q < 2; ++q) {
      float4 v = p[q], s = ps[q], t = pb[q];
      float e[4] = { v.x*s.x + t.x, v.y*s.y + t.y, v.z*s.z + t.z, v.w*s.w + t.w };
#pragma unroll
      for (int j = 0; j < 4; ++j) {
        float r = fminf(fmaxf(e[j], 0.0f), 6.0f);
        a[half*8 + q*4 + j] = (bf16_t)r;
      }
    }
  }
  return a;
}

// ---------------- fused WMMA GEMM ------------------------------------------
// C[M,NC] = f(A[M,K]) @ W[K,NC], f = identity or bn+relu6.
// Block: 256 thr = 8 waves, each wave a 16x64 strip; block tile 128 x 64.
// W strip staged in LDS as bf16 column-major with +8-half row padding:
// row stride (K+8)*2B -> 16B aligned, lane bank stride 4 -> conflict-free.
template<int K, int NC, bool APPLY_BN>
__global__ __launch_bounds__(256) void wmma_gemm_kernel(
    const float* __restrict__ A, const float* __restrict__ W,
    const float* __restrict__ sc, const float* __restrict__ sh,
    float* __restrict__ C, int M) {
  constexpr int LDS_K = K + 8;
  __shared__ bf16_t ldsB[64 * LDS_K];

  const int tid = threadIdx.x;
  const int colBase = blockIdx.y * 64;
  for (int i = tid; i < 64 * K; i += 256) {      // k-fast: contiguous LDS writes
    const int n = i / K, k = i - n * K;
    ldsB[n * LDS_K + k] = (bf16_t)W[(size_t)k * NC + colBase + n];
  }
  __syncthreads();

  const int lane = tid & 31;
  const int wave = tid >> 5;
  const int row0 = blockIdx.x * 128 + wave * 16;
  if (row0 >= M) return;                          // whole-wave guard (M % 16 == 0)

  v8f acc[4] = {};
#pragma unroll
  for (int ks = 0; ks < K / 32; ++ks) {
    const int kk = ks * 32;
    v16bf a;
    if constexpr (APPLY_BN) a = load_a_frag_bn(A, K, sc, sh, row0, kk, lane);
    else                    a = load_a_frag(A, K, row0, kk, lane);
    const int kb = kk + ((lane >> 4) << 3);       // K 8-group per lane group (table order)
#pragma unroll
    for (int j = 0; j < 4; ++j) {
      const bf16_t* bp = &ldsB[(j * 16 + (lane & 15)) * LDS_K + kb];
      v8bf lo = *reinterpret_cast<const v8bf*>(bp);       // ds_load_b128
      v8bf hi = *reinterpret_cast<const v8bf*>(bp + 16);  // ds_load_b128
      v16bf b = __builtin_shufflevector(lo, hi, 0,1,2,3,4,5,6,7,
                                                8,9,10,11,12,13,14,15);
      acc[j] = __builtin_amdgcn_wmma_f32_16x16x32_bf16(false, a, false, b,
                                                       (short)0, acc[j],
                                                       false, false);
    }
  }

  const int rbase = row0 + ((lane >> 4) << 3);
#pragma unroll
  for (int j = 0; j < 4; ++j) {
    const int col = colBase + j * 16 + (lane & 15);
#pragma unroll
    for (int r = 0; r < 8; ++r) C[(size_t)(rbase + r) * NC + col] = acc[j][r];
  }
}

// ---------------- depthwise gather conv (bn1+relu6 folded) -----------------
// 384 threads = one full 1536B row per access -> perfectly coalesced gathers.
// Mask/index loads are wave-uniform -> scalar branch skips missing entries.
__global__ __launch_bounds__(HID) void dwconv_kernel(const float* __restrict__ H1,
                                                     const int* __restrict__ nidx,
                                                     const unsigned char* __restrict__ nmask,
                                                     const float* __restrict__ W2,
                                                     const float* __restrict__ sc1,
                                                     const float* __restrict__ sh1,
                                                     float* __restrict__ H2, int Mrows) {
  const int c = threadIdx.x;
  float wk[KVOL];
#pragma unroll
  for (int k = 0; k < KVOL; ++k) wk[k] = W2[k * HID + c];
  const float s = sc1[c], t = sh1[c];
  const int row0 = blockIdx.x * 8;
#pragma unroll
  for (int rr = 0; rr < 8; ++rr) {
    const int row = row0 + rr;
    if (row >= Mrows) return;
    float acc = 0.0f;
#pragma unroll
    for (int k = 0; k < KVOL; ++k) {
      if (nmask[row * KVOL + k]) {
        const int idx = nidx[row * KVOL + k];
        float v = H1[(size_t)idx * HID + c];
        v = fminf(fmaxf(v * s + t, 0.0f), 6.0f);
        acc += v * wk[k];
      }
    }
    H2[(size_t)row * HID + c] = acc;
  }
}

// ---------------- column sum / sumsq reduction (BN batch stats) ------------
// blockDim.x must be a multiple of C. stats[0..C-1]=sum, stats[C..2C-1]=sumsq.
__global__ void colstats_kernel(const float* __restrict__ src, int Mrows, int C,
                                int rowsPerBlock, float* __restrict__ stats) {
  const int c    = threadIdx.x % C;
  const int sub  = threadIdx.x / C;
  const int nsub = blockDim.x / C;
  const int r0 = blockIdx.x * rowsPerBlock;
  int r1 = r0 + rowsPerBlock; if (r1 > Mrows) r1 = Mrows;
  float s = 0.0f, s2 = 0.0f;
  for (int r = r0 + sub; r < r1; r += nsub) {
    const float v = src[(size_t)r * C + c];
    s += v; s2 += v * v;
  }
  atomicAdd(&stats[c], s);
  atomicAdd(&stats[C + c], s2);
}

__global__ void bnfinalize_kernel(const float* __restrict__ stats,
                                  const float* __restrict__ gamma,
                                  const float* __restrict__ beta,
                                  int C, float invN,
                                  float* __restrict__ scale,
                                  float* __restrict__ shift) {
  const int c = threadIdx.x;
  if (c < C) {
    const float mean = stats[c] * invN;
    const float var  = stats[C + c] * invN - mean * mean;
    const float sc   = gamma[c] * rsqrtf(var + BN_EPS);
    scale[c] = sc;
    shift[c] = beta[c] - mean * sc;
  }
}

__global__ void zero_kernel(float* __restrict__ p, int n) {
  const int i = blockIdx.x * blockDim.x + threadIdx.x;
  if (i < n) p[i] = 0.0f;
}

// ---------------- bn3 + residual -------------------------------------------
__global__ void residual_kernel(const float* __restrict__ Y, const float* __restrict__ X,
                                const float* __restrict__ sc3, const float* __restrict__ sh3,
                                float* __restrict__ out, int total4) {
  const int i = blockIdx.x * blockDim.x + threadIdx.x;
  if (i >= total4) return;
  const float4 y = reinterpret_cast<const float4*>(Y)[i];
  const float4 x = reinterpret_cast<const float4*>(X)[i];
  const int cb = i & (COUT / 4 - 1);          // channel group (COUT/4 = 16)
  const float4 s = reinterpret_cast<const float4*>(sc3)[cb];
  const float4 t = reinterpret_cast<const float4*>(sh3)[cb];
  float4 o;
  o.x = y.x * s.x + t.x + x.x;
  o.y = y.y * s.y + t.y + x.y;
  o.z = y.z * s.z + t.z + x.z;
  o.w = y.w * s.w + t.w + x.w;
  reinterpret_cast<float4*>(out)[i] = o;
}

// ---------------------------------------------------------------------------
extern "C" void kernel_launch(void* const* d_in, const int* in_sizes, int n_in,
                              void* d_out, int out_size, void* d_ws, size_t ws_size,
                              hipStream_t stream) {
  const float* x    = (const float*)d_in[0];
  const float* w1   = (const float*)d_in[1];
  const float* g1   = (const float*)d_in[2];
  const float* b1   = (const float*)d_in[3];
  const float* w2   = (const float*)d_in[4];
  const float* g2   = (const float*)d_in[5];
  const float* b2   = (const float*)d_in[6];
  const float* w3   = (const float*)d_in[7];
  const float* g3   = (const float*)d_in[8];
  const float* b3   = (const float*)d_in[9];
  const int*   nidx = (const int*)d_in[10];
  const unsigned char* nmask = (const unsigned char*)d_in[11];
  float* out = (float*)d_out;

  const int Mrows = in_sizes[0] / CIN;               // 200000
  const float invN = 1.0f / (float)Mrows;

  // Workspace layout (floats). Needs 2*M*HID + ~4K floats (~615 MB).
  const size_t nh = (size_t)Mrows * HID;
  float* ws    = (float*)d_ws;
  float* h1    = ws;                                 // [M,HID]; reused as yraw
  float* h2    = ws + nh;                            // [M,HID]
  float* yraw  = h1;                                 // h1 dead after dwconv
  float* st    = ws + 2 * nh;
  float* st1 = st;               float* st2 = st1 + 2 * HID;
  float* st3 = st2 + 2 * HID;
  float* sc1 = st3 + 2 * COUT;   float* sh1 = sc1 + HID;
  float* sc2 = sh1 + HID;        float* sh2 = sc2 + HID;
  float* sc3 = sh2 + HID;        float* sh3 = sc3 + COUT;

  const int statsN = 2 * HID + 2 * HID + 2 * COUT;   // 1664
  const int CHUNKS = 500;
  const int rowsPerChunk = (Mrows + CHUNKS - 1) / CHUNKS;
  const int mBlocks = (Mrows + 127) / 128;           // 1563

  // 1) zero stats
  zero_kernel<<<(statsN + 255) / 256, 256, 0, stream>>>(st, statsN);
  // 2) GEMM1 (WMMA)
  wmma_gemm_kernel<CIN, HID, false><<<dim3(mBlocks, HID / 64), 256, 0, stream>>>(
      x, w1, nullptr, nullptr, h1, Mrows);
  // 3) BN1 stats + finalize
  colstats_kernel<<<CHUNKS, HID, 0, stream>>>(h1, Mrows, HID, rowsPerChunk, st1);
  bnfinalize_kernel<<<1, HID, 0, stream>>>(st1, g1, b1, HID, invN, sc1, sh1);
  // 4) depthwise gather (bn1+relu6 folded)
  dwconv_kernel<<<(Mrows + 7) / 8, HID, 0, stream>>>(h1, nidx, nmask, w2, sc1, sh1,
                                                     h2, Mrows);
  // 5) BN2 stats + finalize
  colstats_kernel<<<CHUNKS, HID, 0, stream>>>(h2, Mrows, HID, rowsPerChunk, st2);
  bnfinalize_kernel<<<1, HID, 0, stream>>>(st2, g2, b2, HID, invN, sc2, sh2);
  // 6) GEMM3 (WMMA, bn2+relu6 folded into A loads)
  wmma_gemm_kernel<HID, COUT, true><<<dim3(mBlocks, 1), 256, 0, stream>>>(
      h2, w3, sc2, sh2, yraw, Mrows);
  // 7) BN3 stats + finalize
  colstats_kernel<<<CHUNKS, 256, 0, stream>>>(yraw, Mrows, COUT, rowsPerChunk, st3);
  bnfinalize_kernel<<<1, COUT, 0, stream>>>(st3, g3, b3, COUT, invN, sc3, sh3);
  // 8) bn3 + residual
  const int total4 = Mrows * (COUT / 4);
  residual_kernel<<<(total4 + 255) / 256, 256, 0, stream>>>(yraw, x, sc3, sh3, out,
                                                            total4);
}
